// NeuralLongTermMemory_15848429322885
// MI455X (gfx1250) — matrix operand order repaired and checked by reading, not compiled, not run on hardware
//
#include <hip/hip_runtime.h>
#include <hip/hip_bf16.h>
#include <cstdint>
#include <cstddef>

// Problem constants (from reference): B=4, S=4096, D=1024, H=2048
#define BS_ 16384
#define D_  1024
#define H_  2048

typedef __bf16 bf16_t;
typedef bf16_t v16bf __attribute__((ext_vector_type(16)));
typedef float  v8f   __attribute__((ext_vector_type(8)));
typedef unsigned int u32x4 __attribute__((ext_vector_type(4)));

union FragU { v16bf v; u32x4 q[2]; };

__device__ __forceinline__ unsigned short f2bf(float f) {
  unsigned int u = __builtin_bit_cast(unsigned int, f);
  u += 0x7FFFu + ((u >> 16) & 1u);              // round-to-nearest-even
  return (unsigned short)(u >> 16);
}
__device__ __forceinline__ float bf2f(unsigned short h) {
  unsigned int u = ((unsigned int)h) << 16;
  return __builtin_bit_cast(float, u);
}

// ---------------------------------------------------------------- utilities
__global__ void zero_scalars(float* sc) { if (threadIdx.x < 16) sc[threadIdx.x] = 0.f; }

__global__ void cvt_f32_bf16(const float* __restrict__ in, unsigned short* __restrict__ out, size_t n) {
  size_t i  = blockIdx.x * (size_t)blockDim.x + threadIdx.x;
  size_t st = (size_t)gridDim.x * blockDim.x;
  for (; i < n; i += st) out[i] = f2bf(in[i]);
}

// bf16 tile transpose: out(C,R) = in(R,C)^T.  R,C multiples of 32.
__global__ __launch_bounds__(256) void transpose_bf16(const unsigned short* __restrict__ in,
                                                      unsigned short* __restrict__ out,
                                                      int R, int C) {
  __shared__ unsigned short t[32][33];
  int c0 = blockIdx.x * 32, r0 = blockIdx.y * 32;
  int tx = threadIdx.x & 31, ty = threadIdx.x >> 5;
  for (int i = ty; i < 32; i += 8) t[i][tx] = in[(size_t)(r0 + i) * C + c0 + tx];
  __syncthreads();
  for (int i = ty; i < 32; i += 8) out[(size_t)(c0 + i) * R + r0 + tx] = t[tx][i];
}

// ------------------------------------------------------- WMMA fragment loads
// A fragment 16x32 (MxK): lane L -> row m0+(L&15); K chunks [h*8, h*8+8) and
// [16+h*8, 16+h*8+8) with h = L>>4  (ISA 7.12.2, 16-bit A layout).
__device__ __forceinline__ v16bf load_a(const unsigned short* __restrict__ A, int lda, int m0, int k0) {
  int lane = threadIdx.x & 31;
  const unsigned short* p = A + (size_t)(m0 + (lane & 15)) * lda + k0 + (lane >> 4) * 8;
  FragU u;
  u.q[0] = *reinterpret_cast<const u32x4*>(p);
  u.q[1] = *reinterpret_cast<const u32x4*>(p + 16);
  return u.v;
}
// B fragment 32x16 (KxN), sourced from W stored (N,K) row-major (B = W^T):
// lane L -> column n0+(L&15) = row of W; K chunks [h*8, h*8+8) and [16+h*8, 16+h*8+8), h = L>>4.
__device__ __forceinline__ v16bf load_b(const unsigned short* __restrict__ Bm, int ldb, int n0, int k0) {
  int lane = threadIdx.x & 31;
  const unsigned short* p = Bm + (size_t)(n0 + (lane & 15)) * ldb + k0 + (lane >> 4) * 8;
  FragU u;
  u.q[0] = *reinterpret_cast<const u32x4*>(p);
  u.q[1] = *reinterpret_cast<const u32x4*>(p + 16);
  return u.v;
}

// ----------------------------------------------------------------- GEMM core
enum { EP_BF16 = 0, EP_F32, EP_SILU, EP_SILU_BOTH, EP_DPRED, EP_DSILU, EP_GATE };

// C(M,N) = epilogue( A(M,K) @ B(N,K)^T ).  Block: 256 threads = 8 waves,
// block tile 128x128, wave tile 32x64 (2x4 fragments of 16x16).
// M % 128 == 0, N % 128 == 0, K % 32 == 0 (true for all call sites).
template <int EP>
__global__ __launch_bounds__(256)
void wmma_gemm(const unsigned short* __restrict__ A, const unsigned short* __restrict__ Bm,
               int M, int N, int K,
               void* __restrict__ Cp, void* __restrict__ C2p,
               const unsigned short* __restrict__ aux, const float* __restrict__ bias,
               float* __restrict__ part, float scale) {
  const int w    = threadIdx.x >> 5;
  const int lane = threadIdx.x & 31;
  const int wm   = blockIdx.y * 128 + (w >> 1) * 32;
  const int wn   = blockIdx.x * 128 + (w & 1) * 64;

  v8f zero;
  for (int e = 0; e < 8; ++e) zero[e] = 0.f;
  v8f acc[2][4];
  for (int mi = 0; mi < 2; ++mi)
    for (int nj = 0; nj < 4; ++nj) acc[mi][nj] = zero;

  for (int k0 = 0; k0 < K; k0 += 32) {
    v16bf a[2], b[4];
    a[0] = load_a(A, K, wm, k0);
    a[1] = load_a(A, K, wm + 16, k0);
#pragma unroll
    for (int nj = 0; nj < 4; ++nj) b[nj] = load_b(Bm, K, wn + nj * 16, k0);
#pragma unroll
    for (int mi = 0; mi < 2; ++mi)
#pragma unroll
      for (int nj = 0; nj < 4; ++nj)
        acc[mi][nj] = __builtin_amdgcn_wmma_f32_16x16x32_bf16(
            false, a[mi], false, b[nj], (short)0, acc[mi][nj], false, false);
  }

  // Epilogue. C/D layout (ISA 7.12.2): VGPR r, lane L -> row = r + 8*(L>>4),
  // col = L&15 within the 16x16 fragment.
  const int halfl = lane >> 4;
  const int cb    = lane & 15;
  float gsum = 0.f;
  for (int mi = 0; mi < 2; ++mi) {
    for (int nj = 0; nj < 4; ++nj) {
      const int col  = wn + nj * 16 + cb;
      const int row0 = wm + mi * 16 + halfl * 8;
      v8f c = acc[mi][nj];
      for (int r = 0; r < 8; ++r) {
        const size_t idx = (size_t)(row0 + r) * N + col;
        float v = c[r];
        if constexpr (EP == EP_BF16) {
          ((unsigned short*)Cp)[idx] = f2bf(v);
        } else if constexpr (EP == EP_F32) {
          ((float*)Cp)[idx] = v;
        } else if constexpr (EP == EP_SILU || EP == EP_SILU_BOTH) {
          float sg = 1.f / (1.f + __expf(-v));
          ((unsigned short*)Cp)[idx] = f2bf(v * sg);
          if constexpr (EP == EP_SILU_BOTH) ((unsigned short*)C2p)[idx] = f2bf(v);
        } else if constexpr (EP == EP_DPRED) {
          ((unsigned short*)Cp)[idx] = f2bf(scale * (v - bf2f(aux[idx])));
        } else if constexpr (EP == EP_DSILU) {
          float hr = bf2f(aux[idx]);
          float sg = 1.f / (1.f + __expf(-hr));
          ((unsigned short*)Cp)[idx] = f2bf(v * sg * (1.f + hr * (1.f - sg)));
        } else if constexpr (EP == EP_GATE) {
          gsum += 1.f / (1.f + __expf(-(v + bias[col])));
        }
      }
    }
  }
  if constexpr (EP == EP_GATE) {
    __shared__ float red[256];
    red[threadIdx.x] = gsum;
    __syncthreads();
    for (int s = 128; s > 0; s >>= 1) {
      if (threadIdx.x < s) red[threadIdx.x] += red[threadIdx.x + s];
      __syncthreads();
    }
    if (threadIdx.x == 0) part[blockIdx.y * gridDim.x + blockIdx.x] = red[0];
  }
}

// Deterministic gate reduction: part holds 3 x 1024 block partials.
__global__ __launch_bounds__(256) void gate_finalize(const float* __restrict__ part, float* __restrict__ sc) {
  __shared__ float red[256];
  for (int g = 0; g < 3; ++g) {
    float s = 0.f;
    for (int i = threadIdx.x; i < 1024; i += 256) s += part[g * 1024 + i];
    red[threadIdx.x] = s;
    __syncthreads();
    for (int st = 128; st > 0; st >>= 1) {
      if (threadIdx.x < st) red[threadIdx.x] += red[threadIdx.x + st];
      __syncthreads();
    }
    if (threadIdx.x == 0) sc[g] = red[0] * (1.f / ((float)BS_ * (float)D_));
    __syncthreads();
  }
  if (threadIdx.x == 0) {
    sc[4] = 1.f - sc[0];  // 1 - alpha
    sc[5] = sc[1];        // theta
    sc[6] = sc[2];        // eta
  }
}

// Mn = (1-alpha)*M + eta*S - theta*g   (S1/S2 are zero in the reference init,
// but kept general).  Output bf16, consumed by retrieval GEMMs.
__global__ void update_memory(const float* __restrict__ M, const float* __restrict__ S,
                              const unsigned short* __restrict__ g, unsigned short* __restrict__ Mn,
                              const float* __restrict__ sc, size_t n) {
  float om_a = sc[4], theta = sc[5], eta = sc[6];
  size_t i  = blockIdx.x * (size_t)blockDim.x + threadIdx.x;
  size_t st = (size_t)gridDim.x * blockDim.x;
  for (; i < n; i += st) Mn[i] = f2bf(om_a * M[i] + eta * S[i] - theta * bf2f(g[i]));
}

// ------------------------------------------------------------------- launch
extern "C" void kernel_launch(void* const* d_in, const int* in_sizes, int n_in,
                              void* d_out, int out_size, void* d_ws, size_t ws_size,
                              hipStream_t stream) {
  (void)in_sizes; (void)n_in; (void)out_size; (void)ws_size;
  const int BS = BS_, D = D_, H = H_;

  const float* x    = (const float*)d_in[0];
  const float* Wk   = (const float*)d_in[1];
  const float* Wv   = (const float*)d_in[2];
  const float* Wq   = (const float*)d_in[3];
  const float* Wout = (const float*)d_in[4];
  const float* Wgd  = (const float*)d_in[5];
  const float* bgd  = (const float*)d_in[6];
  const float* Wgl  = (const float*)d_in[7];
  const float* bgl  = (const float*)d_in[8];
  const float* Wgm  = (const float*)d_in[9];
  const float* bgm  = (const float*)d_in[10];
  const float* M1   = (const float*)d_in[11];
  const float* M2   = (const float*)d_in[12];
  const float* S1   = (const float*)d_in[13];
  const float* S2   = (const float*)d_in[14];

  char* wsp = (char*)d_ws;
  size_t off = 0;
  auto alloc = [&](size_t bytes) -> char* {
    char* p = wsp + off;
    off += (bytes + 255) & ~(size_t)255;
    return p;
  };
  const size_t szBD = (size_t)BS * D * 2;  // bf16 (BS,D)
  const size_t szBH = (size_t)BS * H * 2;  // bf16 (BS,H)
  const size_t szDD = (size_t)D * D * 2;
  const size_t szHD = (size_t)H * D * 2;

  float* sc   = (float*)alloc(1024);
  float* part = (float*)alloc(3 * 1024 * sizeof(float));
  unsigned short* xb    = (unsigned short*)alloc(szBD);
  unsigned short* wkb   = (unsigned short*)alloc(szDD);
  unsigned short* wvb   = (unsigned short*)alloc(szDD);
  unsigned short* wqb   = (unsigned short*)alloc(szDD);
  unsigned short* woutb = (unsigned short*)alloc(szDD);
  unsigned short* wgdb  = (unsigned short*)alloc(szDD);
  unsigned short* wglb  = (unsigned short*)alloc(szDD);
  unsigned short* wgmb  = (unsigned short*)alloc(szDD);
  unsigned short* m1b   = (unsigned short*)alloc(szHD);
  unsigned short* m2b   = (unsigned short*)alloc(szHD);
  unsigned short* m2tb  = (unsigned short*)alloc(szHD);
  unsigned short* g1b   = (unsigned short*)alloc(szHD);
  unsigned short* g2b   = (unsigned short*)alloc(szHD);
  unsigned short* m1nb  = (unsigned short*)alloc(szHD);
  unsigned short* m2nb  = (unsigned short*)alloc(szHD);
  unsigned short* kb    = (unsigned short*)alloc(szBD);
  unsigned short* vb    = (unsigned short*)alloc(szBD);
  unsigned short* qb    = (unsigned short*)alloc(szBD);
  unsigned short* hs    = (unsigned short*)alloc(szBH);
  unsigned short* hraw  = (unsigned short*)alloc(szBH);
  unsigned short* dh    = (unsigned short*)alloc(szBH);
  unsigned short* dpred = (unsigned short*)alloc(szBD);
  unsigned short* kT     = (unsigned short*)alloc(szBD);
  unsigned short* dpredT = (unsigned short*)alloc(szBD);
  unsigned short* hsT    = (unsigned short*)alloc(szBH);
  unsigned short* dhT    = (unsigned short*)alloc(szBH);
  unsigned short* h2s = hs;     // reuse: hs dead after hsT built
  unsigned short* rb  = dpred;  // reuse: dpred dead after dpredT built

  zero_scalars<<<1, 32, 0, stream>>>(sc);

  auto cvt = [&](const float* in, unsigned short* out, size_t n) {
    cvt_f32_bf16<<<4096, 256, 0, stream>>>(in, out, n);
  };
  cvt(x, xb, (size_t)BS * D);
  cvt(Wk, wkb, (size_t)D * D);   cvt(Wv, wvb, (size_t)D * D);
  cvt(Wq, wqb, (size_t)D * D);   cvt(Wout, woutb, (size_t)D * D);
  cvt(Wgd, wgdb, (size_t)D * D); cvt(Wgl, wglb, (size_t)D * D);
  cvt(Wgm, wgmb, (size_t)D * D);
  cvt(M1, m1b, (size_t)H * D);   cvt(M2, m2b, (size_t)H * D);

  auto grid = [](int M, int N) { return dim3(N / 128, M / 128); };

  // k, v, q projections
  wmma_gemm<EP_BF16><<<grid(BS, D), 256, 0, stream>>>(xb, wkb, BS, D, D, kb, nullptr, nullptr, nullptr, nullptr, 0.f);
  wmma_gemm<EP_BF16><<<grid(BS, D), 256, 0, stream>>>(xb, wvb, BS, D, D, vb, nullptr, nullptr, nullptr, nullptr, 0.f);
  wmma_gemm<EP_BF16><<<grid(BS, D), 256, 0, stream>>>(xb, wqb, BS, D, D, qb, nullptr, nullptr, nullptr, nullptr, 0.f);

  // gate scalars: mean(sigmoid(x @ Wg^T + b)) -> 1024 block partials each
  wmma_gemm<EP_GATE><<<grid(BS, D), 256, 0, stream>>>(xb, wgdb, BS, D, D, nullptr, nullptr, nullptr, bgd, part + 0 * 1024, 0.f);
  wmma_gemm<EP_GATE><<<grid(BS, D), 256, 0, stream>>>(xb, wglb, BS, D, D, nullptr, nullptr, nullptr, bgl, part + 1 * 1024, 0.f);
  wmma_gemm<EP_GATE><<<grid(BS, D), 256, 0, stream>>>(xb, wgmb, BS, D, D, nullptr, nullptr, nullptr, bgm, part + 2 * 1024, 0.f);
  gate_finalize<<<1, 256, 0, stream>>>(part, sc);

  // forward: h = k @ M1^T, keep silu(h) and raw h
  wmma_gemm<EP_SILU_BOTH><<<grid(BS, H), 256, 0, stream>>>(kb, m1b, BS, H, D, hs, hraw, nullptr, nullptr, nullptr, 0.f);
  transpose_bf16<<<dim3(H / 32, BS / 32), 256, 0, stream>>>(hs, hsT, BS, H);

  // dpred = (2/D) * (silu(h) @ M2^T - v)
  wmma_gemm<EP_DPRED><<<grid(BS, D), 256, 0, stream>>>(hs, m2b, BS, D, H, dpred, nullptr, vb, nullptr, nullptr, 2.f / (float)D);

  // dh = (dpred @ M2) * silu'(h)   (needs M2^T stored (H,D))
  transpose_bf16<<<dim3(H / 32, D / 32), 256, 0, stream>>>(m2b, m2tb, D, H);
  wmma_gemm<EP_DSILU><<<grid(BS, H), 256, 0, stream>>>(dpred, m2tb, BS, H, D, dh, nullptr, hraw, nullptr, nullptr, 0.f);

  // transposes for the A^T B gradient GEMMs
  transpose_bf16<<<dim3(D / 32, BS / 32), 256, 0, stream>>>(dpred, dpredT, BS, D);
  transpose_bf16<<<dim3(H / 32, BS / 32), 256, 0, stream>>>(dh, dhT, BS, H);
  transpose_bf16<<<dim3(D / 32, BS / 32), 256, 0, stream>>>(kb, kT, BS, D);

  // g2 = dpred^T @ silu(h)  -> (D,H);  g1 = dh^T @ k -> (H,D)
  wmma_gemm<EP_BF16><<<grid(D, H), 256, 0, stream>>>(dpredT, hsT, D, H, BS, g2b, nullptr, nullptr, nullptr, nullptr, 0.f);
  wmma_gemm<EP_BF16><<<grid(H, D), 256, 0, stream>>>(dhT, kT, H, D, BS, g1b, nullptr, nullptr, nullptr, nullptr, 0.f);

  // memory update with gate scalars
  update_memory<<<2048, 256, 0, stream>>>(M1, S1, g1b, m1nb, sc, (size_t)H * D);
  update_memory<<<2048, 256, 0, stream>>>(M2, S2, g2b, m2nb, sc, (size_t)H * D);

  // retrieval: out = (silu(q @ M1n^T) @ M2n^T) @ Wout^T
  wmma_gemm<EP_SILU><<<grid(BS, H), 256, 0, stream>>>(qb, m1nb, BS, H, D, h2s, nullptr, nullptr, nullptr, nullptr, 0.f);
  wmma_gemm<EP_BF16><<<grid(BS, D), 256, 0, stream>>>(h2s, m2nb, BS, D, H, rb, nullptr, nullptr, nullptr, nullptr, 0.f);
  wmma_gemm<EP_F32><<<grid(BS, D), 256, 0, stream>>>(rb, woutb, BS, D, D, (float*)d_out, nullptr, nullptr, nullptr, nullptr, 0.f);
}
